// MS_Linear_Attention_59373627899909
// MI455X (gfx1250) — matrix operand order repaired and checked by reading, not compiled, not run on hardware
//
#include <hip/hip_runtime.h>
#include <hip/hip_bf16.h>
#include <cstdint>

typedef __attribute__((ext_vector_type(16))) _Float16 v16h;
typedef __attribute__((ext_vector_type(8)))  _Float16 v8h;
typedef __attribute__((ext_vector_type(8)))  float    v8f;
typedef __attribute__((ext_vector_type(4)))  unsigned int v4u;
typedef __attribute__((ext_vector_type(8)))  int      v8i;
typedef __attribute__((ext_vector_type(4)))  int      v4i;

#define DEVFN __device__ __forceinline__

DEVFN v8f wmma16(v16h a, v16h b, v8f c) {
  // v_wmma_f32_16x16x32_f16  (neg_a, A, neg_b, B, c_mod, C, reuse_a, reuse_b)
  return __builtin_amdgcn_wmma_f32_16x16x32_f16(false, a, false, b, (short)0, c, false, false);
}

DEVFN float elu1(float x)  { return x > 0.f ? x + 1.f : __expf(x); }
DEVFN float lrelu(float x) { return x > 0.f ? x : 0.2f * x; }

constexpr int NSP = 4096;     // H*W
constexpr int CCH = 256;      // C
constexpr float ATT_EPS = 1e-6f;

// ---------------------------------------------------------------------------
// x (B,256,4096) f32 -> xT (B,4096,256) f16, LDS-tiled 32x32 transpose.
// ---------------------------------------------------------------------------
__global__ __launch_bounds__(256) void transpose_cvt_kernel(
    const float* __restrict__ in, _Float16* __restrict__ out) {
  __shared__ float tile[32][33];
  const int b = blockIdx.z, c0 = blockIdx.y * 32, n0 = blockIdx.x * 32;
  const int tx = threadIdx.x & 31, ty0 = threadIdx.x >> 5;
#pragma unroll
  for (int i = 0; i < 4; ++i) {
    const int ty = ty0 + i * 8;
    tile[ty][tx] = in[((size_t)(b * 256 + c0 + ty)) * 4096 + n0 + tx];
  }
  __syncthreads();
#pragma unroll
  for (int i = 0; i < 4; ++i) {
    const int ty = ty0 + i * 8;
    out[((size_t)(b * 4096 + n0 + ty)) * 256 + c0 + tx] = (_Float16)tile[tx][ty];
  }
}

// ---------------------------------------------------------------------------
// Weight repack: W (Cout,Cin) f32 -> fragment-major f16.
// out[(((ot*(Cin/32)+kb)*32+lane)*16)+e] = W[ot*16+(lane&15)][kb*32+16*(lane>>4)+e]
// => A operand for a k-step is ONE contiguous 32B v16h load per lane.
// ---------------------------------------------------------------------------
__global__ __launch_bounds__(256) void repack_wfrag_kernel(
    const float* __restrict__ w, _Float16* __restrict__ out, int Cin, int Cout) {
  const int idx = blockIdx.x * 256 + threadIdx.x;
  if (idx >= Cout * Cin) return;
  const int e = idx & 15, lane = (idx >> 4) & 31, blk = idx >> 9;
  const int kblocks = Cin >> 5;
  const int kb = blk % kblocks, ot = blk / kblocks;
  const int row = ot * 16 + (lane & 15);
  const int col = kb * 32 + 16 * (lane >> 4) + e;
  out[idx] = (_Float16)w[(size_t)row * Cin + col];
}

// Wg2 (64,64,3,3) -> tap-major (k = t*64+ci) + fragment-major f16.
__global__ __launch_bounds__(256) void repack_wg2frag_kernel(
    const float* __restrict__ w, _Float16* __restrict__ out) {
  const int idx = blockIdx.x * 256 + threadIdx.x;  // < 64*576
  const int e = idx & 15, lane = (idx >> 4) & 31, blk = idx >> 9;
  const int kb = blk % 18, ot = blk / 18;
  const int co = ot * 16 + (lane & 15);
  const int k  = kb * 32 + 16 * (lane >> 4) + e;
  const int t = k >> 6, ci = k & 63;
  out[idx] = (_Float16)w[(size_t)(co * 64 + ci) * 9 + t];
}

// ---------------------------------------------------------------------------
// 1x1-conv GEMM, all-f16 operands, fragment-major weights:
//   X:(B,4096,Cin) f16, Wf:(frag-major) f16 -> act(W@X + bias)
// One wave: 16(out-ch) x 64(spatial) tile, K-steps of 32.
// ACT: 0 none, 1 leaky(0.2), 2 elu+1
// OUTM: 0 f32 (B,Cout,N) | 1 f16 (B,N,Cout) packed v8h | 2 f16 (B,Cout,N)
// ---------------------------------------------------------------------------
template <int ACT, int OUTM>
__global__ __launch_bounds__(256) void gemm_f16_kernel(
    const _Float16* __restrict__ X, const _Float16* __restrict__ W,
    const float* __restrict__ bias, void* __restrict__ Yv, int Cin, int Cout) {
  const int lane = threadIdx.x & 31;
  const int lhi  = lane >> 4, l15 = lane & 15;
  const int tile = blockIdx.x * 8 + (threadIdx.x >> 5);
  const int nt   = tile & 63;               // NSP/64 == 64 spatial tiles
  const int rest = tile >> 6;
  const int otCount = Cout >> 4;
  const int ot = rest % otCount;
  const int b  = rest / otCount;
  const int n0 = nt * 64;
  const _Float16* Xb = X + (size_t)b * NSP * Cin + (size_t)(n0 + l15) * Cin;

  v8f acc0 = {}, acc1 = {}, acc2 = {}, acc3 = {};

  for (int k0 = 0; k0 < Cin; k0 += 32) {
    const v16h a = *(const v16h*)(
        W + ((((size_t)ot * (Cin >> 5) + (k0 >> 5)) * 32 + lane) << 4));
    const _Float16* xrow = Xb + k0 + 16 * lhi;
    acc0 = wmma16(a, *(const v16h*)(xrow), acc0);
    acc1 = wmma16(a, *(const v16h*)(xrow + (size_t)16 * Cin), acc1);
    acc2 = wmma16(a, *(const v16h*)(xrow + (size_t)32 * Cin), acc2);
    acc3 = wmma16(a, *(const v16h*)(xrow + (size_t)48 * Cin), acc3);
  }

  const v8f accs[4] = {acc0, acc1, acc2, acc3};
#pragma unroll
  for (int sub = 0; sub < 4; ++sub) {
    if constexpr (OUTM == 1) {
      v8h pack;
#pragma unroll
      for (int r = 0; r < 8; ++r) {
        const int row = ot * 16 + 8 * lhi + r;
        float v = accs[sub][r];
        if (bias) v += bias[row];
        if constexpr (ACT == 1) v = lrelu(v);
        if constexpr (ACT == 2) v = elu1(v);
        pack[r] = (_Float16)v;
      }
      _Float16* Y = (_Float16*)Yv;
      *(v8h*)(Y + ((size_t)(b * NSP + n0 + sub * 16 + l15)) * Cout + ot * 16 + 8 * lhi) = pack;
    } else {
#pragma unroll
      for (int r = 0; r < 8; ++r) {
        const int row = ot * 16 + 8 * lhi + r;
        float v = accs[sub][r];
        if (bias) v += bias[row];
        if constexpr (ACT == 1) v = lrelu(v);
        if constexpr (ACT == 2) v = elu1(v);
        const size_t off = (size_t)(b * Cout + row) * NSP + n0 + sub * 16 + l15;
        if constexpr (OUTM == 0) ((float*)Yv)[off] = v;
        else                     ((_Float16*)Yv)[off] = (_Float16)v;
      }
    }
  }
}

// ---------------------------------------------------------------------------
// Dense 3x3 conv (64->64), implicit-im2col WMMA GEMM, tap-major K = 9*64.
// One tap per 16-wide fragment: single bounds check + one 32B load.
// ---------------------------------------------------------------------------
__global__ __launch_bounds__(256) void gemm_conv3x3_kernel(
    const _Float16* __restrict__ X, const _Float16* __restrict__ W,
    const float* __restrict__ bias, _Float16* __restrict__ Y) {
  constexpr int Cin = 64, Cout = 64, KF = 576;
  const int lane = threadIdx.x & 31;
  const int lhi  = lane >> 4, l15 = lane & 15;
  const int tile = blockIdx.x * 8 + (threadIdx.x >> 5);
  const int nt   = tile & 63;
  const int rest = tile >> 6;
  const int ot = rest % (Cout >> 4);
  const int b  = rest / (Cout >> 4);
  const int n0 = nt * 64;
  const _Float16* Xb = X + (size_t)b * NSP * Cin;

  v8f acc0 = {}, acc1 = {}, acc2 = {}, acc3 = {};

  for (int k0 = 0; k0 < KF; k0 += 32) {
    const v16h a = *(const v16h*)(
        W + ((((size_t)ot * 18 + (k0 >> 5)) * 32 + lane) << 4));

    const int base = k0 + 16 * lhi;       // never crosses a 64-boundary
    const int t = base >> 6;              // tap id 0..8
    const int cib = base & 63;            // channel run start
    const int dy = t / 3 - 1, dx = t - (t / 3) * 3 - 1;

    v8f* accp[4] = {&acc0, &acc1, &acc2, &acc3};
#pragma unroll
    for (int sub = 0; sub < 4; ++sub) {
      const int n = n0 + sub * 16 + l15;
      const int yy = (n >> 6) + dy, xx = (n & 63) + dx;
      v16h bf;
      if (yy >= 0 && yy < 64 && xx >= 0 && xx < 64) {
        bf = *(const v16h*)(Xb + ((size_t)((yy << 6) + xx)) * Cin + cib);
      } else {
#pragma unroll
        for (int e = 0; e < 16; ++e) bf[e] = (_Float16)0.f;
      }
      *accp[sub] = wmma16(a, bf, *accp[sub]);
    }
  }

  const v8f accs[4] = {acc0, acc1, acc2, acc3};
#pragma unroll
  for (int sub = 0; sub < 4; ++sub) {
    v8h pack;
#pragma unroll
    for (int r = 0; r < 8; ++r) {
      const int row = ot * 16 + 8 * lhi + r;
      pack[r] = (_Float16)lrelu(accs[sub][r] + bias[row]);
    }
    *(v8h*)(Y + ((size_t)(b * NSP + n0 + sub * 16 + l15)) * Cout + ot * 16 + 8 * lhi) = pack;
  }
}

// ---------------------------------------------------------------------------
// Depthwise 3x3 over 512 channels, f16 in/out, f32 math; elu+1 on k-half.
// ---------------------------------------------------------------------------
__global__ __launch_bounds__(256) void dwconv_kernel(
    const _Float16* __restrict__ A, const float* __restrict__ Wdw,
    _Float16* __restrict__ Out) {
  const int idx = blockIdx.x * 256 + threadIdx.x;   // (b*512+c)*4096+pix
  const int pix = idx & 4095;
  const int c2  = (idx >> 12) & 511;
  const int y = pix >> 6, x = pix & 63;
  const _Float16* in = A + (size_t)(idx - pix);
  const float* w = Wdw + c2 * 9;
  float s = 0.f;
#pragma unroll
  for (int ky = 0; ky < 3; ++ky) {
#pragma unroll
    for (int kx = 0; kx < 3; ++kx) {
      const int yy = y + ky - 1, xx = x + kx - 1;
      if (yy >= 0 && yy < 64 && xx >= 0 && xx < 64)
        s += w[ky * 3 + kx] * (float)in[(yy << 6) + xx];
    }
  }
  if (c2 < 256) s = elu1(s);
  Out[idx] = (_Float16)s;
}

// ---------------------------------------------------------------------------
// Chunked causal linear attention, one wave per (batch, head). Chunk T=16.
// State S (64 x 80) f32 in LDS; col 64 = cumsum(k). V augmented with a
// ones-column tile so the denominator falls out of the same WMMAs.
// K chunk (16 pos x 64 feat, stride N) is DMA'd into LDS by the Tensor Data
// Mover each chunk (tensor_load_to_lds + s_wait_tensorcnt); reversed heads
// use reversed column indexing into the TDM tile.
// ---------------------------------------------------------------------------
__global__ __launch_bounds__(32) void attn_kernel(
    const _Float16* __restrict__ qT, const _Float16* __restrict__ kv,
    _Float16* __restrict__ oT) {
  constexpr int N = NSP, C = CCH, D = 64;
  const int bh = blockIdx.x;
  const int b = bh >> 2, h = bh & 3;
  const bool rev = (h >= 2);
  const int lane = threadIdx.x;
  const int lhi = lane >> 4, l15 = lane & 15;

  const _Float16* kb = kv + (size_t)(b * 2 * C + h * D) * N;          // k (B,C,N)
  const _Float16* vb = kv + (size_t)(b * 2 * C + C + h * D) * N;      // v (B,C,N)
  _Float16* ob = oT + (size_t)b * N * C + h * D;

  __shared__ float    S32[64][81];                    // f32 state (cols 0..79)
  __shared__ __align__(64) _Float16 K16d[64][16];     // TDM tile: [d][pos], 32B rows
  __shared__ _Float16 P16[16][40];                    // masked P (cols>=16 zero)

  for (int i = lane; i < 64 * 81; i += 32) ((float*)S32)[i] = 0.f;
  for (int i = lane; i < 16 * 40; i += 32) ((_Float16*)P16)[i] = (_Float16)0.f;
  __syncthreads();

  const unsigned int k16_lds = (unsigned int)(uintptr_t)(&K16d[0][0]);
  const int jcol = rev ? (15 - l15) : l15;            // logical pos -> tile col

  for (int chk = 0; chk < N / 16; ++chk) {
    const int n0 = chk * 16;
    const int pstart = rev ? (N - 16 - n0) : n0;

    // ---- issue TDM: K tile (16 pos x 64 d) global -> LDS [d][pos] ----
    {
      const unsigned long long ga =
          (unsigned long long)(uintptr_t)kb + (unsigned long long)pstart * 2u;
      v4u g0;
      g0[0] = 1u;                                   // count=1, user descriptor
      g0[1] = k16_lds;                              // lds_addr
      g0[2] = (unsigned int)ga;                     // global_addr[31:0]
      g0[3] = (unsigned int)(ga >> 32) | (2u << 30);// global_addr[56:32] | type=2
      v8i g1;
      g1[0] = 0x10000;        // data_size=2B
      g1[1] = 0x10000000;     // tensor_dim0 = 4096  (bits 79:48)
      g1[2] = 64 << 16;       // tensor_dim1 = 64    (bits 111:80)
      g1[3] = 16 << 16;       // tile_dim0   = 16    (bits 127:112)
      g1[4] = 64;             // tile_dim1   = 64
      g1[5] = 4096;           // tensor_dim0_stride
      g1[6] = 0; g1[7] = 0;
      const v4i gz4 = {0, 0, 0, 0};
      const v8i gz8 = {0, 0, 0, 0, 0, 0, 0, 0};
      __builtin_amdgcn_tensor_load_to_lds(g0, g1, gz4, gz4, gz8, 0);
    }

    // ---- Q A-fragments: wide loads from qT (B,N,256) ----
    const int pq = rev ? (N - 1 - (n0 + l15)) : (n0 + l15);
    const _Float16* qrow = qT + ((size_t)(b * N + pq)) * C + h * D;
    v16h aQ[2];
#pragma unroll
    for (int ks = 0; ks < 2; ++ks) {
      const v8h lo = *(const v8h*)(qrow + ks * 32 + 16 * lhi);
      const v8h hi = *(const v8h*)(qrow + ks * 32 + 16 * lhi + 8);
#pragma unroll
      for (int e = 0; e < 8; ++e) { aQ[ks][e] = lo[e]; aQ[ks][8 + e] = hi[e]; }
    }

    // ---- O_inter = Q @ S_prev (5 tiles incl. denominator tile) ----
    v8f accO[5] = {v8f{}, v8f{}, v8f{}, v8f{}, v8f{}};
#pragma unroll
    for (int et = 0; et < 5; ++et) {
#pragma unroll
      for (int ks = 0; ks < 2; ++ks) {
        v16h bS;
#pragma unroll
        for (int e = 0; e < 16; ++e)
          bS[e] = (_Float16)S32[ks * 32 + 16 * lhi + e][et * 16 + l15];
        accO[et] = wmma16(aQ[ks], bS, accO[et]);
      }
    }

    // ---- wait for TDM tile, then P = Q @ K^T ----
    __builtin_amdgcn_s_wait_tensorcnt(0);
    v8f accP = {};
#pragma unroll
    for (int ks = 0; ks < 2; ++ks) {
      v16h bK;
#pragma unroll
      for (int e = 0; e < 16; ++e)
        bK[e] = K16d[ks * 32 + 16 * lhi + e][jcol];
      accP = wmma16(aQ[ks], bK, accP);
    }
    // ---- causal mask (incl. diagonal) -> f16 A operand ----
#pragma unroll
    for (int r = 0; r < 8; ++r) {
      const int m = 8 * lhi + r;
      P16[m][l15] = (_Float16)((l15 <= m) ? accP[r] : 0.f);
    }
    __syncthreads();
    v16h aP;
    {
      const v8h plo = *(const v8h*)&P16[l15][8 * lhi];
      const v8h phi = *(const v8h*)&P16[l15][8 * lhi + 16];  // zeros
#pragma unroll
      for (int e = 0; e < 8; ++e) { aP[e] = plo[e]; aP[8 + e] = phi[e]; }
    }

    // ---- V B-fragments, wide loads (tile 4 = ones column) ----
    v16h bV[5];
#pragma unroll
    for (int et = 0; et < 4; ++et) {
      v16h bf;
      {
        const v8h wv = *(const v8h*)(vb + ((size_t)(et * 16 + l15)) * N + pstart +
                                     (rev ? 8 * (1 - lhi) : 8 * lhi));
#pragma unroll
        for (int e = 0; e < 8; ++e) {
          bf[e] = rev ? wv[7 - e] : wv[e];
          bf[8 + e] = (_Float16)0.f;
        }
      }
      bV[et] = bf;
    }
#pragma unroll
    for (int e = 0; e < 16; ++e)
      bV[4][e] = (_Float16)((l15 == 0 && e < 8) ? 1.0f : 0.0f);

    // ---- O_intra = maskP @ V ----
#pragma unroll
    for (int et = 0; et < 5; ++et) accO[et] = wmma16(aP, bV[et], accO[et]);

    // ---- scale by 1/(den+eps), den = col0 of tile 4 ----
    float scale[8];
#pragma unroll
    for (int r = 0; r < 8; ++r) {
      const int den_i = __builtin_amdgcn_ds_bpermute((lane & 16) * 4,
                                                     __float_as_int(accO[4][r]));
      scale[r] = 1.0f / (__int_as_float(den_i) + ATT_EPS);
    }
#pragma unroll
    for (int et = 0; et < 4; ++et) {
#pragma unroll
      for (int r = 0; r < 8; ++r) {
        const int m = 8 * lhi + r;
        const int p = rev ? (N - 1 - (n0 + m)) : (n0 + m);
        ob[(size_t)p * C + et * 16 + l15] = (_Float16)(accO[et][r] * scale[r]);
      }
    }

    // ---- state update: S += K^T @ V_aug (f32 accumulate in LDS) ----
#pragma unroll
    for (int mt = 0; mt < 4; ++mt) {
      // A = K^T tile: lane row d = mt*16+l15; k-elems = chunk positions.
      const v8h kr = *(const v8h*)&K16d[mt * 16 + l15][rev ? 8 * (1 - lhi) : 8 * lhi];
      v16h aK;
#pragma unroll
      for (int e = 0; e < 8; ++e) {
        aK[e] = rev ? kr[7 - e] : kr[e];
        aK[8 + e] = (_Float16)0.f;
      }
#pragma unroll
      for (int et = 0; et < 5; ++et) {
        v8f ct;
#pragma unroll
        for (int r = 0; r < 8; ++r)
          ct[r] = S32[mt * 16 + 8 * lhi + r][et * 16 + l15];
        ct = wmma16(aK, bV[et], ct);
#pragma unroll
        for (int r = 0; r < 8; ++r)
          S32[mt * 16 + 8 * lhi + r][et * 16 + l15] = ct[r];
      }
    }
    __syncthreads();
  }
}

// ---------------------------------------------------------------------------
// Gating multiply: both operands in (B,N,C) f16 -> gated (B,N,C) f16.
// ---------------------------------------------------------------------------
__global__ __launch_bounds__(256) void gatemul_kernel(
    const _Float16* __restrict__ g3, const _Float16* __restrict__ o,
    _Float16* __restrict__ out) {
  const int idx = blockIdx.x * 256 + threadIdx.x;
  out[idx] = (_Float16)((float)g3[idx] * (float)o[idx]);
}

// ---------------------------------------------------------------------------
extern "C" void kernel_launch(void* const* d_in, const int* in_sizes, int n_in,
                              void* d_out, int out_size, void* d_ws,
                              size_t ws_size, hipStream_t stream) {
  const float* x   = (const float*)d_in[0];
  const float* Wq  = (const float*)d_in[1];
  const float* bq  = (const float*)d_in[2];
  const float* Wa1 = (const float*)d_in[3];
  const float* Wdw = (const float*)d_in[4];
  const float* Wo  = (const float*)d_in[5];
  const float* bo  = (const float*)d_in[6];
  const float* Wg1 = (const float*)d_in[7];
  const float* bg1 = (const float*)d_in[8];
  const float* Wg2 = (const float*)d_in[9];
  const float* bg2 = (const float*)d_in[10];
  const float* Wg3 = (const float*)d_in[11];
  const float* bg3 = (const float*)d_in[12];
  float* out = (float*)d_out;
  char* ws = (char*)d_ws;

  // f16 workspace buffers (byte offsets, all >=256B aligned)
  _Float16* xT     = (_Float16*)(ws);                    // 2*4096*256
  _Float16* qTb    = (_Float16*)(ws + (4u << 20));       // 2*4096*256
  _Float16* a16    = (_Float16*)(ws + (8u << 20));       // 2*512*4096
  _Float16* kv16   = (_Float16*)(ws + (16u << 20));      // 2*512*4096
  _Float16* oTb    = (_Float16*)(ws + (24u << 20));      // 2*4096*256
  _Float16* g1T    = (_Float16*)(ws + (28u << 20));      // 2*4096*64
  _Float16* g2T    = (_Float16*)(ws + (29u << 20));      // 2*4096*64
  _Float16* g3T    = (_Float16*)(ws + (30u << 20));      // 2*4096*256
  _Float16* gatedT = (_Float16*)(ws + (34u << 20));      // 2*4096*256
  _Float16* w16q   = (_Float16*)(ws + (38u << 20));      // 65536
  _Float16* w16a1  = (_Float16*)(ws + (38u << 20) + 131072);   // 131072
  _Float16* w16g1  = (_Float16*)(ws + (38u << 20) + 393216);   // 16384
  _Float16* w16g2t = (_Float16*)(ws + (38u << 20) + 425984);   // 36864
  _Float16* w16g3  = (_Float16*)(ws + (38u << 20) + 499712);   // 16384
  _Float16* w16o   = (_Float16*)(ws + (38u << 20) + 532480);   // 65536

  const dim3 blk(256);

  // ---- one-time operand repack to f16 (fragment-major weights) ----
  transpose_cvt_kernel<<<dim3(128, 8, 2), blk, 0, stream>>>(x, xT);
  repack_wfrag_kernel<<<65536 / 256, blk, 0, stream>>>(Wq, w16q, 256, 256);
  repack_wfrag_kernel<<<131072 / 256, blk, 0, stream>>>(Wa1, w16a1, 256, 512);
  repack_wfrag_kernel<<<16384 / 256, blk, 0, stream>>>(Wg1, w16g1, 256, 64);
  repack_wfrag_kernel<<<16384 / 256, blk, 0, stream>>>(Wg3, w16g3, 64, 256);
  repack_wfrag_kernel<<<65536 / 256, blk, 0, stream>>>(Wo, w16o, 256, 256);
  repack_wg2frag_kernel<<<36864 / 256, blk, 0, stream>>>(Wg2, w16g2t);

  // 1) q = elu(conv1x1(x,Wq)+bq)+1 -> qT (B,N,C) f16
  gemm_f16_kernel<2, 1><<<(2 * 16 * 64) / 8, blk, 0, stream>>>(xT, w16q, bq, qTb, 256, 256);
  // 2) a = conv1x1(x, Wa1) -> (B,512,N) f16
  gemm_f16_kernel<0, 2><<<(2 * 32 * 64) / 8, blk, 0, stream>>>(xT, w16a1, nullptr, a16, 256, 512);
  // 3) kv = dwconv3x3(a, Wdw); elu+1 fused on k-half
  dwconv_kernel<<<(2 * 512 * 4096) / 256, blk, 0, stream>>>(a16, Wdw, kv16);
  // 4) chunked causal linear attention (TDM-staged K) -> oT (B,N,C) f16
  attn_kernel<<<8, dim3(32), 0, stream>>>(qTb, kv16, oTb);
  // 5) g1 = leaky(conv1x1(x, Wg1)+bg1) -> (B,N,64) f16
  gemm_f16_kernel<1, 1><<<(2 * 4 * 64) / 8, blk, 0, stream>>>(xT, w16g1, bg1, g1T, 256, 64);
  // 6) g2 = leaky(conv3x3(g1, Wg2)+bg2), tap-major implicit im2col
  gemm_conv3x3_kernel<<<(2 * 4 * 64) / 8, blk, 0, stream>>>(g1T, w16g2t, bg2, g2T);
  // 7) g3 = conv1x1(g2, Wg3)+bg3 -> (B,N,256) f16
  gemm_f16_kernel<0, 1><<<(2 * 16 * 64) / 8, blk, 0, stream>>>(g2T, w16g3, bg3, g3T, 64, 256);
  // 8) gated = g3 * attention-out (same layout, pure elementwise)
  gatemul_kernel<<<(2 * 4096 * 256) / 256, blk, 0, stream>>>(g3T, oTb, gatedT);
  // 9) out = conv1x1(gated, Wo)+bo -> f32 (B,C,H,W)
  gemm_f16_kernel<0, 0><<<(2 * 16 * 64) / 8, blk, 0, stream>>>(gatedT, w16o, bo, out, 256, 256);
}
